// AutoModelForMatch_52132313038852
// MI455X (gfx1250) — matrix operand order repaired and checked by reading, not compiled, not run on hardware
//
#include <hip/hip_runtime.h>
#include <hip/hip_bf16.h>

typedef _Float16 v16h __attribute__((ext_vector_type(16)));
typedef _Float16 v8h  __attribute__((ext_vector_type(8)));
typedef float    v8f  __attribute__((ext_vector_type(8)));

#define Q_TOTAL  2048
#define N_TOTAL  262144
#define D_DIM    768
#define TOPK     100
#define PBLOCKS  8
#define QB       64                      // queries per workgroup (shared by all 4 waves)
#define QTILES   4                       // 4 query tiles of 16 per wave step
#define WAVES    4
#define NSLICE   (N_TOTAL / PBLOCKS)     // 32768 passages per WG slice
#define WTILES   (NSLICE / 16 / WAVES)   // 512 passage tiles per wave
#define NLISTS   (PBLOCKS * WAVES)       // 32 partial lists per query
#define LCAND    (NLISTS * TOPK)         // 3200 candidates per query
#define MPAD     4096                    // bitonic pad
#define A_STRIDE 776                     // padded LDS row stride (halves)
#define NEG_INF  (-3.402823466e38f)

// ---------------------------------------------------------------------------
// Kernel 1: fused f16-WMMA GEMM + streaming per-wave top-k.
// Block = 128 threads (4 wave32). 64 query rows staged once in LDS (f16) and
// shared by all waves; each wave owns a disjoint passage sub-range, so one
// B-fragment load/convert feeds FOUR v_wmma ops (4 query tiles x 1 passage
// tile). No barriers in the hot loop; top-k lists are per-wave.
// ---------------------------------------------------------------------------
__global__ __launch_bounds__(128)
void gemm_topk_partial(const float* __restrict__ query,
                       const float* __restrict__ passage,
                       float* __restrict__ pvals,
                       int*   __restrict__ pinds)
{
    extern __shared__ char smem[];
    _Float16* Aq   = (_Float16*)smem;                               // QB * A_STRIDE halves
    float*    topV = (float*)(smem + (size_t)QB * A_STRIDE * 2);    // WAVES*QB*TOPK
    int*      topI = (int*)  ((char*)topV + (size_t)WAVES * QB * TOPK * 4);
    float*    thr  = (float*)((char*)topI + (size_t)WAVES * QB * TOPK * 4);
    int*      mpos = (int*)  ((char*)thr  + (size_t)WAVES * QB * 4);

    const int tid    = threadIdx.x;
    const int lane   = tid & 31;
    const int wave   = tid >> 5;
    const int qBase  = blockIdx.y * QB;
    const int pb     = blockIdx.x;
    const int pSlice = pb * NSLICE;

    // init per-wave top-k state
    for (int i = tid; i < WAVES * QB * TOPK; i += 128) { topV[i] = NEG_INF; topI[i] = 0; }
    for (int i = tid; i < WAVES * QB; i += 128)        { thr[i]  = NEG_INF; mpos[i] = 0; }

    // stage 64 query rows as f16 into LDS (coalesced fp32 reads + convert)
    for (int i = tid; i < QB * D_DIM; i += 128) {
        int qq = i / D_DIM, kx = i - qq * D_DIM;
        Aq[qq * A_STRIDE + kx] = (_Float16)query[(size_t)(qBase + qq) * D_DIM + kx];
    }
    __syncthreads();

    const int m    = lane & 15;   // A row within tile / B column (passage) for this lane
    const int hi   = lane >> 4;
    const int kbA  = hi * 16;     // A-matrix K offset per lane half (contiguous, matches B)
    const int kbB  = hi * 16;     // B-matrix K offset per lane half
    const int LB   = wave * QB;   // this wave's list base

    for (int tile = 0; tile < WTILES; ++tile) {
        const int pBase = pSlice + (wave * WTILES + tile) * 16;
        const float* brow = passage + (size_t)(pBase + m) * D_DIM;
        __builtin_prefetch(brow + (size_t)16 * D_DIM, 0, 1);   // next tile, same lane-row

        v8f acc[QTILES];
        #pragma unroll
        for (int qt = 0; qt < QTILES; ++qt) acc[qt] = (v8f){};

        #pragma unroll 2
        for (int kk = 0; kk < D_DIM; kk += 32) {
            // B fragment: lane = passage column, 16 consecutive K, f32 -> f16
            const float* bp = brow + kk + kbB;
            float4 f0 = *(const float4*)(bp + 0);
            float4 f1 = *(const float4*)(bp + 4);
            float4 f2 = *(const float4*)(bp + 8);
            float4 f3 = *(const float4*)(bp + 12);
            v16h b;
            b[0]=(_Float16)f0.x;  b[1]=(_Float16)f0.y;  b[2]=(_Float16)f0.z;  b[3]=(_Float16)f0.w;
            b[4]=(_Float16)f1.x;  b[5]=(_Float16)f1.y;  b[6]=(_Float16)f1.z;  b[7]=(_Float16)f1.w;
            b[8]=(_Float16)f2.x;  b[9]=(_Float16)f2.y;  b[10]=(_Float16)f2.z; b[11]=(_Float16)f2.w;
            b[12]=(_Float16)f3.x; b[13]=(_Float16)f3.y; b[14]=(_Float16)f3.z; b[15]=(_Float16)f3.w;

            // one B fragment feeds 4 WMMAs (4 shared-A query tiles)
            #pragma unroll
            for (int qt = 0; qt < QTILES; ++qt) {
                const _Float16* arow = &Aq[(qt * 16 + m) * A_STRIDE + kk];
                v8h alo = *(const v8h*)(arow + kbA);
                v8h ahi = *(const v8h*)(arow + 8 + kbA);
                v16h a = __builtin_shufflevector(alo, ahi,
                         0,1,2,3,4,5,6,7,8,9,10,11,12,13,14,15);
                acc[qt] = __builtin_amdgcn_wmma_f32_16x16x32_f16(
                              false, a, false, b, (short)0, acc[qt], false, false);
            }
        }

        // streaming top-k update. C layout: VGPR r, lanes0-15 -> M=r, N=lane;
        // lanes16-31 -> M=8+r, N=lane-16.
        #pragma unroll
        for (int qt = 0; qt < QTILES; ++qt) {
            #pragma unroll
            for (int r = 0; r < 8; ++r) {
                float v = acc[qt][r];
                v = (v == v) ? v : 0.0f;               // nan_to_num
                const int ql   = qt * 16 + r + 8 * hi; // local query index (0..63)
                const int pidx = pBase + m;            // passage index
                const int qoff = (LB + ql) * TOPK;
                bool cand = v > thr[LB + ql];
                unsigned long long bal = __ballot(cand);
                if (bal) {
                    for (int l = 0; l < 32; ++l) {
                        if (lane == l) {
                            if (v > thr[LB + ql]) {    // re-check: threshold may rise
                                int mp = mpos[LB + ql];
                                topV[qoff + mp] = v;
                                topI[qoff + mp] = pidx;
                                float mn = topV[qoff]; int best = 0;
                                for (int j = 1; j < TOPK; ++j) {
                                    float t2 = topV[qoff + j];
                                    if (t2 < mn) { mn = t2; best = j; }
                                }
                                mpos[LB + ql] = best;
                                thr[LB + ql]  = mn;
                            }
                        }
                    }
                }
            }
        }
    }

    // write this wave's 64 partial lists (list id = pb*WAVES + wave)
    const int list = pb * WAVES + wave;
    for (int t = lane; t < QB * TOPK; t += 32) {
        int ql = t / TOPK, j = t - ql * TOPK;
        size_t o = (size_t)(qBase + ql) * LCAND + (size_t)list * TOPK + j;
        pvals[o] = topV[(LB + ql) * TOPK + j];
        pinds[o] = topI[(LB + ql) * TOPK + j];
    }
}

// ---------------------------------------------------------------------------
// Kernel 2: merge 32 partial top-100 lists per query via bitonic sort (desc),
// emit sorted top-100 (indices + values).
// ---------------------------------------------------------------------------
__global__ __launch_bounds__(256)
void merge_topk(const float* __restrict__ pvals, const int* __restrict__ pinds,
                int* __restrict__ outI, float* __restrict__ outV)
{
    __shared__ float sv[MPAD];
    __shared__ int   si[MPAD];
    const int q   = blockIdx.x;
    const int tid = threadIdx.x;

    for (int i = tid; i < MPAD; i += 256) {
        if (i < LCAND) {
            sv[i] = pvals[(size_t)q * LCAND + i];
            si[i] = pinds[(size_t)q * LCAND + i];
        } else {
            sv[i] = NEG_INF; si[i] = 0x7fffffff;
        }
    }
    __syncthreads();

    for (int k = 2; k <= MPAD; k <<= 1) {
        for (int j = k >> 1; j > 0; j >>= 1) {
            for (int i = tid; i < MPAD; i += 256) {
                int ix = i ^ j;
                if (ix > i) {
                    bool descSeg = ((i & k) == 0);
                    float a = sv[i], b = sv[ix];
                    bool sw = descSeg ? (a < b) : (a > b);
                    if (sw) {
                        sv[i] = b; sv[ix] = a;
                        int t = si[i]; si[i] = si[ix]; si[ix] = t;
                    }
                }
            }
            __syncthreads();
        }
    }

    for (int i = tid; i < TOPK; i += 256) {
        outI[(size_t)q * TOPK + i] = si[i];
        outV[(size_t)q * TOPK + i] = sv[i];
    }
}

// ---------------------------------------------------------------------------
extern "C" void kernel_launch(void* const* d_in, const int* in_sizes, int n_in,
                              void* d_out, int out_size, void* d_ws, size_t ws_size,
                              hipStream_t stream) {
    const float* query   = (const float*)d_in[0];
    const float* passage = (const float*)d_in[1];

    // workspace: partial top-k lists (values then indices): 2048*32*100 each
    float* pvals = (float*)d_ws;
    int*   pinds = (int*)((char*)d_ws + (size_t)Q_TOTAL * LCAND * sizeof(float));

    // outputs: (indices, values) concatenated flat in return order
    int*   outI = (int*)d_out;
    float* outV = (float*)d_out + (size_t)Q_TOTAL * TOPK;

    const size_t smem = (size_t)QB * A_STRIDE * 2          // staged f16 queries (97 KB)
                      + (size_t)WAVES * QB * TOPK * 8      // per-wave top-k vals+idx (200 KB)
                      + (size_t)WAVES * QB * 8;            // thresholds + min positions
    (void)hipFuncSetAttribute(reinterpret_cast<const void*>(gemm_topk_partial),
                              hipFuncAttributeMaxDynamicSharedMemorySize, (int)smem);

    dim3 g1(PBLOCKS, Q_TOTAL / QB);   // 8 x 32 = 256 workgroups
    gemm_topk_partial<<<g1, 128, smem, stream>>>(query, passage, pvals, pinds);
    merge_topk<<<Q_TOTAL, 256, 0, stream>>>(pvals, pinds, outI, outV);
}
